// VisionEncoder_29575144800937
// MI455X (gfx1250) — compile-verified, matrix operand order checked by reading, not run on hardware
//
#include <hip/hip_runtime.h>

// ---------------------------------------------------------------------------
// CDNA5 (gfx1250) wave32 bf16-WMMA vision encoder.
// GEMMs: v_wmma_f32_16x16x32_bf16, block tile 128x64 (8 waves as 4x2 strips,
// wave tile 32x32; A and B fragments each reused across two WMMAs), A+B
// staged in LDS from b128 global loads with register-prefetch pipeline.
// Attention: flash-style, one wave per (head, 16-query tile).
// ---------------------------------------------------------------------------

typedef __bf16 bf16;
typedef __attribute__((ext_vector_type(16))) __bf16 v16bf;
typedef __attribute__((ext_vector_type(8)))  float  v8f;

#define NTOK   4096
#define DMODEL 1024
#define NH     16
#define HDIM   64
#define FFDIM  4096
#define SEG1   1024   // end of image 0 tokens
#define SEG2   2560   // end of image 1 tokens
#define KPATCH 768    // 3*16*16

union Frag {
  v16bf    v;
  bf16     h[16];
  unsigned u[8];
};

static __device__ __forceinline__ v8f wmma_bf16(const Frag& a, const Frag& b, v8f c) {
  return __builtin_amdgcn_wmma_f32_16x16x32_bf16(false, a.v, false, b.v,
                                                 (short)0, c, false, false);
}

// ---------------------------------------------------------------------------
// GEMM: C[M,N] = A[M,K]*B[K,N], A/B bf16, f32 accumulate.
// 256 threads = 8 waves (4 M-strips x 2 N-strips). Block tile 128x64,
// wave tile 32x32, K-step 32.
//   sA : 128 rows x 16 pairs, pitch 20 dwords (16B-aligned, conflict-free)
//   sBT: B tile transposed, 64 n-rows x 32 k, pitch 34 u16 (aligned pairs)
// Per K-step per wave: 2 A frags + 2 B frags -> 4 WMMAs (both reused 2x).
// Register prefetch kept small (3 uint4) to avoid scratch spill.
// Epilogue: f32 out (+residual) and/or bf16 out (optionally transposed).
// ---------------------------------------------------------------------------
__global__ __launch_bounds__(256)
void gemm_bf16_kernel(const bf16* __restrict__ A, const bf16* __restrict__ B,
                      float* __restrict__ Cf, bf16* __restrict__ Cb,
                      const float* __restrict__ Res,
                      int M, int N, int K, int cbT)
{
  __shared__ unsigned       sA[128 * 20];
  __shared__ unsigned short sBT[64 * 34];

  const int tid  = threadIdx.x;
  const int lane = tid & 31;
  const int wave = tid >> 5;
  const int mw   = wave & 3;      // M strip (4 x 32 rows)
  const int nw   = wave >> 2;     // N strip (2 x 32 cols)
  const int half = lane >> 4;
  const int l15  = lane & 15;
  const int m0   = blockIdx.y * 128;
  const int n0   = blockIdx.x * 64;

  // cooperative-load geometry
  const int arow0 = tid >> 2;     // A: rows tid>>2 and tid>>2+64, 4 uint4/row
  const int aquad = tid & 3;
  const int brow  = tid >> 3;     // B: 32 rows, 8 x uint4 per row
  const int bquad = tid & 7;

  uint4 aR0, aR1, bReg;
  auto loadTile = [&](int k0) {
    aR0  = *(const uint4*)(A + (size_t)(m0 + arow0) * K + k0 + aquad * 8);
    aR1  = *(const uint4*)(A + (size_t)(m0 + arow0 + 64) * K + k0 + aquad * 8);
    bReg = *(const uint4*)(B + (size_t)(k0 + brow) * N + n0 + bquad * 8);
  };
  auto storeTile = [&]() {
    *(uint4*)&sA[(size_t)arow0 * 20 + aquad * 4]        = aR0;
    *(uint4*)&sA[(size_t)(arow0 + 64) * 20 + aquad * 4] = aR1;
    // unpack B dwords by value (no address-of on registers)
    unsigned w0 = bReg.x, w1 = bReg.y, w2 = bReg.z, w3 = bReg.w;
    const int nb = bquad * 8;
    sBT[(size_t)(nb + 0) * 34 + brow] = (unsigned short)(w0 & 0xffff);
    sBT[(size_t)(nb + 1) * 34 + brow] = (unsigned short)(w0 >> 16);
    sBT[(size_t)(nb + 2) * 34 + brow] = (unsigned short)(w1 & 0xffff);
    sBT[(size_t)(nb + 3) * 34 + brow] = (unsigned short)(w1 >> 16);
    sBT[(size_t)(nb + 4) * 34 + brow] = (unsigned short)(w2 & 0xffff);
    sBT[(size_t)(nb + 5) * 34 + brow] = (unsigned short)(w2 >> 16);
    sBT[(size_t)(nb + 6) * 34 + brow] = (unsigned short)(w3 & 0xffff);
    sBT[(size_t)(nb + 7) * 34 + brow] = (unsigned short)(w3 >> 16);
  };

  v8f acc[2][2] = {};   // [M-subtile][N-chunk]
  loadTile(0);

  for (int k0 = 0; k0 < K; k0 += 32) {
    storeTile();
    __syncthreads();
    if (k0 + 32 < K) loadTile(k0 + 32);   // prefetch next tile (overlaps WMMA)

    Frag af0, af1;
    const int r0 = mw * 32 + l15;
#pragma unroll
    for (int j = 0; j < 8; ++j) {
      int pi = (j >> 2) * 8 + half * 4 + (j & 3);   // K-pair index (ISA layout)
      af0.u[j] = sA[r0 * 20 + pi];
      af1.u[j] = sA[(r0 + 16) * 20 + pi];
    }
#pragma unroll
    for (int c = 0; c < 2; ++c) {
      Frag bfr;
      const int n = nw * 32 + c * 16 + l15;
#pragma unroll
      for (int j = 0; j < 8; ++j) {
        int kk = (j >> 2) * 16 + half * 8 + (j & 3) * 2;
        bfr.u[j] = *(const unsigned*)&sBT[n * 34 + kk];
      }
      acc[0][c] = wmma_bf16(af0, bfr, acc[0][c]);   // B fragment reused twice
      acc[1][c] = wmma_bf16(af1, bfr, acc[1][c]);
    }
    __syncthreads();
  }

  // Epilogue: C layout — VGPR j: lanes 0-15 row j, lanes 16-31 row j+8.
#pragma unroll
  for (int sub = 0; sub < 2; ++sub) {
#pragma unroll
    for (int c = 0; c < 2; ++c) {
#pragma unroll
      for (int j = 0; j < 8; ++j) {
        int row = m0 + mw * 32 + sub * 16 + (half ? (j + 8) : j);
        int col = n0 + nw * 32 + c * 16 + l15;
        size_t idx = (size_t)row * N + col;
        float v = acc[sub][c][j];
        if (Res) v += Res[idx];
        if (Cf)  Cf[idx] = v;
        if (Cb) {
          if (cbT) Cb[(size_t)col * M + row] = (bf16)v;   // transposed (for V)
          else     Cb[idx] = (bf16)v;
        }
      }
    }
  }
}

// ---------------------------------------------------------------------------
// Flash attention, one wave per (head, 16-query tile).
// Q,K: bf16 (H, NTOK, HDIM).  V: bf16 transposed (H*HDIM, NTOK).
// ---------------------------------------------------------------------------
__global__ __launch_bounds__(32)
void attn_kernel(const bf16* __restrict__ Qb, const bf16* __restrict__ Kb,
                 const bf16* __restrict__ Vt, bf16* __restrict__ Ob)
{
  __shared__ bf16 sP[16 * 34];   // 16 rows x 32 probs (+2 pad), 68B pitch

  const int h    = blockIdx.y;
  const int qt   = blockIdx.x;
  const int lane = threadIdx.x;
  const int half = lane >> 4;
  const int l15  = lane & 15;
  const int q0   = qt * 16;

  int s0, s1;
  if (q0 < SEG1)      { s0 = 0;    s1 = SEG1; }
  else if (q0 < SEG2) { s0 = SEG1; s1 = SEG2; }
  else                { s0 = SEG2; s1 = NTOK; }

  const float scale = 0.125f;  // 1/sqrt(64)

  // Q fragments: two K-chunks (dims 0-31, 32-63)
  Frag qf[2];
  {
    const bf16* qrow = Qb + ((size_t)h * NTOK + (q0 + l15)) * HDIM;
#pragma unroll
    for (int ch = 0; ch < 2; ++ch)
#pragma unroll
      for (int j = 0; j < 8; ++j) {
        int d0 = ch * 32 + (j >> 2) * 16 + half * 8 + (j & 3) * 2;
        qf[ch].u[j] = *(const unsigned*)(qrow + d0);
      }
  }

  v8f acc[4] = {};
  float mrow[8], lrow[8];
#pragma unroll
  for (int j = 0; j < 8; ++j) { mrow[j] = -1e30f; lrow[j] = 0.f; }

  for (int kb = s0; kb < s1; kb += 32) {
    // ---- scores: two 16-key halves, each K=64 via 2 chained WMMAs ----
    v8f sacc[2];
#pragma unroll
    for (int kk = 0; kk < 2; ++kk) {
      v8f s = {};
      const bf16* krow = Kb + ((size_t)h * NTOK + (kb + kk * 16 + l15)) * HDIM;
#pragma unroll
      for (int ch = 0; ch < 2; ++ch) {
        Frag kf;
#pragma unroll
        for (int j = 0; j < 8; ++j) {
          int d0 = ch * 32 + (j >> 2) * 16 + half * 8 + (j & 3) * 2;
          kf.u[j] = *(const unsigned*)(krow + d0);
        }
        s = wmma_bf16(qf[ch], kf, s);
      }
      sacc[kk] = s;
    }

    // ---- online softmax (row stats reduced inside 16-lane halves) ----
    float pa[8], pb[8], alpha[8];
#pragma unroll
    for (int j = 0; j < 8; ++j) {
      float sa = sacc[0][j] * scale;
      float sb = sacc[1][j] * scale;
      float t = fmaxf(sa, sb);
#pragma unroll
      for (int m = 1; m < 16; m <<= 1) t = fmaxf(t, __shfl_xor(t, m, 32));
      float mn = fmaxf(mrow[j], t);
      float a  = expf(mrow[j] - mn);
      pa[j] = expf(sa - mn);
      pb[j] = expf(sb - mn);
      float rs = pa[j] + pb[j];
#pragma unroll
      for (int m = 1; m < 16; m <<= 1) rs += __shfl_xor(rs, m, 32);
      lrow[j]  = lrow[j] * a + rs;
      mrow[j]  = mn;
      alpha[j] = a;
    }
#pragma unroll
    for (int c = 0; c < 4; ++c)
#pragma unroll
      for (int j = 0; j < 8; ++j) acc[c][j] *= alpha[j];

    // ---- re-lay P (C-layout) into LDS as 16x32 bf16 (A-layout source) ----
    __syncthreads();
#pragma unroll
    for (int j = 0; j < 8; ++j) {
      int r = half ? (j + 8) : j;
      sP[r * 34 + l15]      = (bf16)pa[j];
      sP[r * 34 + 16 + l15] = (bf16)pb[j];
    }
    __syncthreads();

    Frag pf;
#pragma unroll
    for (int j = 0; j < 8; ++j) {
      int pi = (j >> 2) * 8 + half * 4 + (j & 3);
      pf.u[j] = *(const unsigned*)&sP[l15 * 34 + pi * 2];
    }

    // ---- O += P * V : V^T layout makes each fragment pair one dword ----
#pragma unroll
    for (int c = 0; c < 4; ++c) {
      Frag vf;
      const bf16* vrow = Vt + ((size_t)h * HDIM + c * 16 + l15) * NTOK;
#pragma unroll
      for (int j = 0; j < 8; ++j) {
        int key = kb + (j >> 2) * 16 + half * 8 + (j & 3) * 2;
        vf.u[j] = *(const unsigned*)(vrow + key);
      }
      acc[c] = wmma_bf16(pf, vf, acc[c]);
    }
  }

  // ---- normalize and store (token-major, col = h*64 + d) ----
#pragma unroll
  for (int j = 0; j < 8; ++j) {
    float inv = 1.f / lrow[j];
    int row = q0 + (half ? (j + 8) : j);
#pragma unroll
    for (int c = 0; c < 4; ++c) {
      int col = h * HDIM + c * 16 + l15;
      Ob[(size_t)row * DMODEL + col] = (bf16)(acc[c][j] * inv);
    }
  }
}

// ---------------------------------------------------------------------------
// RMSNorm: one block per row of 1024; optional f32 in-place, always bf16 out.
// ---------------------------------------------------------------------------
__global__ __launch_bounds__(256)
void rmsnorm_kernel(float* __restrict__ X, const float* __restrict__ g,
                    bf16* __restrict__ Hb, int writef32)
{
  __shared__ float red[8];
  const int row = blockIdx.x;
  float* x = X + (size_t)row * DMODEL;
  float v[4];
  float ss = 0.f;
#pragma unroll
  for (int i = 0; i < 4; ++i) {
    v[i] = x[threadIdx.x + 256 * i];
    ss += v[i] * v[i];
  }
#pragma unroll
  for (int m = 1; m < 32; m <<= 1) ss += __shfl_xor(ss, m, 32);
  if ((threadIdx.x & 31) == 0) red[threadIdx.x >> 5] = ss;
  __syncthreads();
  float tot = red[0] + red[1] + red[2] + red[3] +
              red[4] + red[5] + red[6] + red[7];
  float rs = rsqrtf(tot * (1.f / DMODEL) + 1e-5f);
#pragma unroll
  for (int i = 0; i < 4; ++i) {
    int c = threadIdx.x + 256 * i;
    float y = v[i] * rs * g[c];
    if (writef32) x[c] = y;
    Hb[(size_t)row * DMODEL + c] = (bf16)y;
  }
}

static __device__ __forceinline__ void tok2rc(int t, int& r, int& c) {
  if (t < SEG1)      { r = t >> 5;           c = t & 31; }
  else if (t < SEG2) { int u = t - SEG1; r = u >> 5; c = u & 31; }
  else               { int u = t - SEG2; r = u / 48; c = u % 48; }
}

// RoPE: rotate q,k pairs; write bf16 in (H, NTOK, HDIM) layout.
__global__ __launch_bounds__(256)
void rope_kernel(const float* __restrict__ Ql, const float* __restrict__ Kl,
                 bf16* __restrict__ Qb, bf16* __restrict__ Kb)
{
  int id = blockIdx.x * 256 + threadIdx.x;
  if (id >= NTOK * NH * 32) return;
  int j  = id & 31;
  int hh = (id >> 5) & 15;
  int t  = id >> 9;
  int r, c;
  tok2rc(t, r, c);
  float ang;
  if (j < 16) ang = (float)r * __powf(10000.f, -(4.f * j) / 64.f);
  else        ang = (float)c * __powf(10000.f, -(4.f * (j - 16) + 2.f) / 64.f);
  float cs = __cosf(ang), sn = __sinf(ang);
  size_t src = (size_t)t * DMODEL + hh * HDIM + 2 * j;
  size_t dst = ((size_t)hh * NTOK + t) * HDIM + 2 * j;
  {
    float q0 = Ql[src], q1 = Ql[src + 1];
    Qb[dst]     = (bf16)(q0 * cs - q1 * sn);
    Qb[dst + 1] = (bf16)(q0 * sn + q1 * cs);
  }
  {
    float k0 = Kl[src], k1 = Kl[src + 1];
    Kb[dst]     = (bf16)(k0 * cs - k1 * sn);
    Kb[dst + 1] = (bf16)(k0 * sn + k1 * cs);
  }
}

// SwiGLU gate: mb = bf16(silu(g) * u)
__global__ __launch_bounds__(256)
void silu_kernel(const bf16* __restrict__ G, const bf16* __restrict__ U,
                 bf16* __restrict__ Mb, int n)
{
  int id = blockIdx.x * 256 + threadIdx.x;
  if (id >= n) return;
  float g = (float)G[id];
  float u = (float)U[id];
  Mb[id] = (bf16)(g / (1.f + expf(-g)) * u);
}

// flat f32 -> bf16
__global__ __launch_bounds__(256)
void cvt_kernel(const float* __restrict__ S, bf16* __restrict__ D, int n)
{
  int id = blockIdx.x * 256 + threadIdx.x;
  if (id < n) D[id] = (bf16)S[id];
}

// conv weights (D, 768) row-major -> (768, D) bf16
__global__ __launch_bounds__(256)
void tconv_kernel(const float* __restrict__ W, bf16* __restrict__ D)
{
  int id = blockIdx.x * 256 + threadIdx.x;
  if (id >= KPATCH * DMODEL) return;
  int n = id & (DMODEL - 1);
  int k = id >> 10;
  D[(size_t)k * DMODEL + n] = (bf16)W[(size_t)n * KPATCH + k];
}

// im2col: token t, k = ch*256 + ph*16 + pw -> bf16 patch matrix (NTOK, 768)
__global__ __launch_bounds__(256)
void im2col_kernel(const float* __restrict__ I0, const float* __restrict__ I1,
                   const float* __restrict__ I2, bf16* __restrict__ A)
{
  int id = blockIdx.x * 256 + threadIdx.x;
  if (id >= NTOK * KPATCH) return;
  int k = id % KPATCH;
  int t = id / KPATCH;
  int ch = k >> 8, ph = (k >> 4) & 15, pw = k & 15;
  int r, c;
  tok2rc(t, r, c);
  const float* img;
  int Hpx, Wpx;
  if (t < SEG1)      { img = I0; Hpx = 512; Wpx = 512; }
  else if (t < SEG2) { img = I1; Hpx = 768; Wpx = 512; }
  else               { img = I2; Hpx = 512; Wpx = 768; }
  A[(size_t)t * KPATCH + k] =
      (bf16)img[(size_t)ch * Hpx * Wpx + (size_t)(r * 16 + ph) * Wpx + (c * 16 + pw)];
}

// ---------------------------------------------------------------------------
// Host orchestration
// ---------------------------------------------------------------------------
extern "C" void kernel_launch(void* const* d_in, const int* in_sizes, int n_in,
                              void* d_out, int out_size, void* d_ws, size_t ws_size,
                              hipStream_t stream)
{
  const float* img0     = (const float*)d_in[0];
  const float* img1     = (const float*)d_in[1];
  const float* img2     = (const float*)d_in[2];
  const float* conv_w   = (const float*)d_in[3];
  const float* ln_pre_w = (const float*)d_in[4];
  const float* ln1_w    = (const float*)d_in[5];
  const float* ln2_w    = (const float*)d_in[6];
  const float* wq       = (const float*)d_in[7];
  const float* wk       = (const float*)d_in[8];
  const float* wv       = (const float*)d_in[9];
  const float* wo       = (const float*)d_in[10];
  const float* w1       = (const float*)d_in[11];
  const float* w3       = (const float*)d_in[12];
  const float* w2       = (const float*)d_in[13];
  float* X = (float*)d_out;   // residual stream lives in d_out (NTOK x D f32)

  char* p = (char*)d_ws;
  auto alloc = [&](size_t bytes) -> void* {
    void* r = (void*)p;
    p += (bytes + 255) & ~(size_t)255;
    return r;
  };

  bf16*  Abf  = (bf16*)alloc((size_t)NTOK * KPATCH * 2);
  bf16*  Wc   = (bf16*)alloc((size_t)KPATCH * DMODEL * 2);
  bf16*  Wqb  = (bf16*)alloc((size_t)2 * DMODEL * DMODEL * 2);
  bf16*  Wkb  = (bf16*)alloc((size_t)2 * DMODEL * DMODEL * 2);
  bf16*  Wvb  = (bf16*)alloc((size_t)2 * DMODEL * DMODEL * 2);
  bf16*  Wob  = (bf16*)alloc((size_t)2 * DMODEL * DMODEL * 2);
  bf16*  W1b  = (bf16*)alloc((size_t)2 * DMODEL * FFDIM * 2);
  bf16*  W3b  = (bf16*)alloc((size_t)2 * DMODEL * FFDIM * 2);
  bf16*  W2b  = (bf16*)alloc((size_t)2 * FFDIM * DMODEL * 2);
  bf16*  hb   = (bf16*)alloc((size_t)NTOK * DMODEL * 2);
  float* qlin = (float*)alloc((size_t)NTOK * DMODEL * 4);
  float* klin = (float*)alloc((size_t)NTOK * DMODEL * 4);
  bf16*  qb   = (bf16*)alloc((size_t)NTOK * DMODEL * 2);
  bf16*  kb   = (bf16*)alloc((size_t)NTOK * DMODEL * 2);
  bf16*  vb   = (bf16*)alloc((size_t)NTOK * DMODEL * 2);   // transposed (H*HD, NTOK)
  bf16*  ob   = (bf16*)alloc((size_t)NTOK * DMODEL * 2);
  bf16*  gb   = (bf16*)alloc((size_t)NTOK * FFDIM * 2);
  bf16*  ub   = (bf16*)alloc((size_t)NTOK * FFDIM * 2);
  bf16*  mb   = (bf16*)alloc((size_t)NTOK * FFDIM * 2);

  auto cvt = [&](const float* s, bf16* d, int n) {
    cvt_kernel<<<dim3((n + 255) / 256), 256, 0, stream>>>(s, d, n);
  };
  auto gemm = [&](const bf16* A, const bf16* B, float* Cf, bf16* Cb,
                  const float* Res, int M, int N, int K, int cbT) {
    gemm_bf16_kernel<<<dim3(N / 64, M / 128), 256, 0, stream>>>(A, B, Cf, Cb, Res, M, N, K, cbT);
  };

  // ---- weight conversion (bf16, L2-resident afterwards) ----
  cvt(wq, Wqb, 2 * DMODEL * DMODEL);
  cvt(wk, Wkb, 2 * DMODEL * DMODEL);
  cvt(wv, Wvb, 2 * DMODEL * DMODEL);
  cvt(wo, Wob, 2 * DMODEL * DMODEL);
  cvt(w1, W1b, 2 * DMODEL * FFDIM);
  cvt(w3, W3b, 2 * DMODEL * FFDIM);
  cvt(w2, W2b, 2 * FFDIM * DMODEL);
  tconv_kernel<<<dim3((KPATCH * DMODEL) / 256), 256, 0, stream>>>(conv_w, Wc);

  // ---- patch embed: im2col + GEMM into d_out, then RMSNorm(ln_pre) ----
  im2col_kernel<<<dim3((NTOK * KPATCH) / 256), 256, 0, stream>>>(img0, img1, img2, Abf);
  gemm(Abf, Wc, X, nullptr, nullptr, NTOK, DMODEL, KPATCH, 0);
  rmsnorm_kernel<<<NTOK, 256, 0, stream>>>(X, ln_pre_w, hb, 1);

  for (int l = 0; l < 2; ++l) {
    const size_t wd = (size_t)l * DMODEL * DMODEL;
    const size_t wf = (size_t)l * DMODEL * FFDIM;

    // attention
    rmsnorm_kernel<<<NTOK, 256, 0, stream>>>(X, ln1_w + l * DMODEL, hb, 0);
    gemm(hb, Wqb + wd, qlin, nullptr, nullptr, NTOK, DMODEL, DMODEL, 0);
    gemm(hb, Wkb + wd, klin, nullptr, nullptr, NTOK, DMODEL, DMODEL, 0);
    gemm(hb, Wvb + wd, nullptr, vb, nullptr, NTOK, DMODEL, DMODEL, 1); // V^T bf16
    rope_kernel<<<dim3((NTOK * NH * 32) / 256), 256, 0, stream>>>(qlin, klin, qb, kb);
    attn_kernel<<<dim3(NTOK / 16, NH), 32, 0, stream>>>(qb, kb, vb, ob);
    gemm(ob, Wob + wd, X, nullptr, X, NTOK, DMODEL, DMODEL, 0);        // + residual

    // MLP (SwiGLU)
    rmsnorm_kernel<<<NTOK, 256, 0, stream>>>(X, ln2_w + l * DMODEL, hb, 0);
    gemm(hb, W1b + wf, nullptr, gb, nullptr, NTOK, FFDIM, DMODEL, 0);
    gemm(hb, W3b + wf, nullptr, ub, nullptr, NTOK, FFDIM, DMODEL, 0);
    silu_kernel<<<dim3((NTOK * FFDIM) / 256), 256, 0, stream>>>(gb, ub, mb, NTOK * FFDIM);
    gemm(mb, W2b + wf, X, nullptr, X, NTOK, DMODEL, FFDIM, 0);         // + residual
  }
}
